// BahdanauAttention_50019189129556
// MI455X (gfx1250) — matrix operand order repaired and checked by reading, not compiled, not run on hardware
//
#include <hip/hip_runtime.h>
#include <hip/hip_bf16.h>
#include <math.h>

#define D_ENC 512
#define D_DEC 512
#define D_ATT 256
#define BATCH 32
#define T_IN  4096

typedef __attribute__((ext_vector_type(16))) __bf16 v16bf;
typedef __attribute__((ext_vector_type(8)))  float  v8f;

union FragU { uint4 u[2]; v16bf v; };

__device__ __forceinline__ unsigned short f32_to_bf16_rne(float f) {
    unsigned int u = __builtin_bit_cast(unsigned int, f);
    unsigned int r = u + 0x7FFFu + ((u >> 16) & 1u);
    return (unsigned short)(r >> 16);
}

// branch-free tanh: 1 - 2/(1 + 2^(2*log2(e)*x)), clamped.
// one v_exp_f32 + one v_rcp_f32, no EXEC divergence.
__device__ __forceinline__ float fast_tanh(float x) {
    float xc = fminf(fmaxf(x, -9.0f), 9.0f);
    float e2 = __builtin_amdgcn_exp2f(xc * 2.8853900817779268f);  // 2*log2(e)
    float r  = __builtin_amdgcn_rcpf(e2 + 1.0f);
    return __builtin_fmaf(-2.0f, r, 1.0f);
}

__device__ __forceinline__ float fast_exp(float x) {
    return __builtin_amdgcn_exp2f(x * 1.4426950408889634f);       // log2(e)
}

// ---------------------------------------------------------------- W_enc -> bf16
__global__ void __launch_bounds__(256)
k_cvt_w(const float* __restrict__ w, unsigned short* __restrict__ o) {
    int i = blockIdx.x * 256 + threadIdx.x;
    if (i < D_ATT * D_ENC) o[i] = f32_to_bf16_rne(w[i]);
}

// ------------------------------------------- dbias[b,a] = dh[b]·W_dec[a] + b_dec + b_enc
__global__ void __launch_bounds__(256)
k_dec_bias(const float* __restrict__ dh, const float* __restrict__ Wdec,
           const float* __restrict__ bdec, const float* __restrict__ benc,
           float* __restrict__ dbias) {
    int b = blockIdx.x;
    int a = threadIdx.x;                       // 0..255
    const float* h = dh + (size_t)b * D_DEC;
    const float* w = Wdec + (size_t)a * D_DEC;
    float s = 0.f;
    #pragma unroll 4
    for (int e = 0; e < D_DEC; e += 4) {
        s += h[e] * w[e] + h[e + 1] * w[e + 1] + h[e + 2] * w[e + 2] + h[e + 3] * w[e + 3];
    }
    dbias[b * D_ATT + a] = s + bdec[a] + benc[a];
}

// ------------------------------------------------------- WMMA scores kernel
// block: 128 threads (4 waves), handles batch b = blockIdx.y, rows [t0, t0+64)
// dynamic LDS: 64 rows x 520 bf16 (1040 B/row, 16B pad for bank spread)
#define LROW 1040   // bytes per LDS row (520 bf16)
__global__ void __launch_bounds__(128)
k_scores(const float* __restrict__ enc, const unsigned char* __restrict__ mask,
         const unsigned short* __restrict__ wencb, const float* __restrict__ dbias,
         const float* __restrict__ vvec, float* __restrict__ scores) {
    extern __shared__ char smem[];
    const int b      = blockIdx.y;
    const int t_base = blockIdx.x * 64;
    const int tid    = threadIdx.x;

    // ---- stage A tile fp32 -> bf16 into LDS (coalesced float4 loads) ----
    const float* A = enc + ((size_t)b * T_IN + t_base) * D_ENC;
    const int c = tid * 4;                     // 128 threads * 4 = 512 cols
    #pragma unroll 4
    for (int row = 0; row < 64; ++row) {
        float4 f = *(const float4*)(A + (size_t)row * D_ENC + c);
        unsigned long long p =
            (unsigned long long)f32_to_bf16_rne(f.x)
          | ((unsigned long long)f32_to_bf16_rne(f.y) << 16)
          | ((unsigned long long)f32_to_bf16_rne(f.z) << 32)
          | ((unsigned long long)f32_to_bf16_rne(f.w) << 48);
        *(unsigned long long*)(smem + (size_t)row * LROW + (size_t)c * 2) = p;
    }
    __syncthreads();

    // ---- compute: each wave owns 16 rows ----
    const int wave    = tid >> 5;
    const int lane    = tid & 31;
    const int l15     = lane & 15;
    const int h       = lane >> 4;             // lane half (selects K +16, contiguous)
    const int m_local = wave * 16;
    const char* arow  = smem + (size_t)(m_local + l15) * LROW;

    float sacc[8];
    #pragma unroll
    for (int r = 0; r < 8; ++r) sacc[r] = 0.f;

    for (int nt = 0; nt < 16; ++nt) {
        const int n0 = nt * 16;
        const float db = dbias[b * D_ATT + n0 + l15];
        const float vv = vvec[n0 + l15];
        const unsigned short* brow = wencb + (size_t)(n0 + l15) * D_ENC;

        v8f acc = {};
        #pragma unroll 4
        for (int k0 = 0; k0 < D_ENC; k0 += 32) {
            FragU af, bf;
            // A frag (16x32 bf16): lane<16 -> K {k0..k0+15}; lane>=16 -> K {k0+16..k0+31}
            af.u[0] = *(const uint4*)(arow + k0 * 2 + h * 32);
            af.u[1] = *(const uint4*)(arow + k0 * 2 + 16 + h * 32);
            // B frag (32x16 bf16): lane n -> column n0+n, K contiguous (h selects +16)
            const uint4* bp = (const uint4*)(brow + k0 + h * 16);
            bf.u[0] = bp[0];
            bf.u[1] = bp[1];
            acc = __builtin_amdgcn_wmma_f32_16x16x32_bf16(
                false, af.v, false, bf.v, (short)0, acc, false, false);
        }
        // fuse: tanh(enc_proj + dbias) · v  (VGPR r = row, lane%16 = column)
        #pragma unroll
        for (int r = 0; r < 8; ++r) sacc[r] += vv * fast_tanh(acc[r] + db);
    }

    // reduce across the 16 columns inside each lane-half, write 16 row scores
    #pragma unroll
    for (int r = 0; r < 8; ++r) {
        float s = sacc[r];
        s += __shfl_xor(s, 1, 32);
        s += __shfl_xor(s, 2, 32);
        s += __shfl_xor(s, 4, 32);
        s += __shfl_xor(s, 8, 32);
        if (l15 == 0) {
            int t = t_base + m_local + r + h * 8;
            size_t idx = (size_t)b * T_IN + t;
            scores[idx] = mask[idx] ? s : -INFINITY;
        }
    }
}

// ------------------------------------------------------- in-place softmax over T
__global__ void __launch_bounds__(256)
k_softmax(float* __restrict__ w) {
    __shared__ float red[256];
    const int b = blockIdx.x;
    const int tid = threadIdx.x;
    float* s = w + (size_t)b * T_IN;

    float m = -INFINITY;
    for (int t = tid; t < T_IN; t += 256) m = fmaxf(m, s[t]);
    red[tid] = m; __syncthreads();
    for (int o = 128; o > 0; o >>= 1) {
        if (tid < o) red[tid] = fmaxf(red[tid], red[tid + o]);
        __syncthreads();
    }
    m = red[0]; __syncthreads();

    float sum = 0.f;
    for (int t = tid; t < T_IN; t += 256) {
        float e = fast_exp(s[t] - m);
        s[t] = e;
        sum += e;
    }
    red[tid] = sum; __syncthreads();
    for (int o = 128; o > 0; o >>= 1) {
        if (tid < o) red[tid] += red[tid + o];
        __syncthreads();
    }
    const float inv = 1.0f / red[0];
    for (int t = tid; t < T_IN; t += 256) s[t] *= inv;
}

// ------------------------------------ context: deterministic 2-stage T reduction
__global__ void __launch_bounds__(256)
k_ctx_partial(const float* __restrict__ enc, const float* __restrict__ w,
              float* __restrict__ partial) {
    const int s   = blockIdx.x;                // t-chunk 0..15
    const int b   = blockIdx.y;
    const int tid = threadIdx.x;
    const float* wb = w + (size_t)b * T_IN + s * 256;
    const float* e0 = enc + ((size_t)b * T_IN + (size_t)s * 256) * D_ENC;
    float a0 = 0.f, a1 = 0.f;
    for (int t = 0; t < 256; ++t) {
        const float wt = wb[t];
        const float* row = e0 + (size_t)t * D_ENC;
        a0 += wt * row[tid];
        a1 += wt * row[tid + 256];
    }
    float* p = partial + ((size_t)s * BATCH + b) * D_ENC;
    p[tid] = a0;
    p[tid + 256] = a1;
}

__global__ void __launch_bounds__(256)
k_ctx_reduce(const float* __restrict__ partial, float* __restrict__ ctx) {
    const int i = blockIdx.x * 256 + threadIdx.x;  // b*512 + e
    float s = 0.f;
    #pragma unroll
    for (int p = 0; p < 16; ++p) s += partial[(size_t)p * (BATCH * D_ENC) + i];
    ctx[i] = s;
}

// ----------------------------------------------------------------------------
extern "C" void kernel_launch(void* const* d_in, const int* in_sizes, int n_in,
                              void* d_out, int out_size, void* d_ws, size_t ws_size,
                              hipStream_t stream) {
    (void)in_sizes; (void)n_in; (void)out_size; (void)ws_size;
    const float*         dh   = (const float*)d_in[0];          // (32, 512)
    const float*         enc  = (const float*)d_in[1];          // (32, 4096, 512)
    const unsigned char* mask = (const unsigned char*)d_in[2];  // (32, 4096) bool
    const float*         Wenc = (const float*)d_in[3];          // (256, 512)
    const float*         benc = (const float*)d_in[4];          // (256,)
    const float*         Wdec = (const float*)d_in[5];          // (256, 512)
    const float*         bdec = (const float*)d_in[6];          // (256,)
    const float*         vvec = (const float*)d_in[7];          // (256,)

    float* out = (float*)d_out;
    float* ctx = out;                          // (32, 512)
    float* wts = out + BATCH * D_ENC;          // (32, 4096)

    char* ws = (char*)d_ws;
    unsigned short* wencb = (unsigned short*)ws;               // 256 KiB bf16 W_enc
    float* dbias   = (float*)(ws + 262144);                    // 32 KiB
    float* partial = (float*)(ws + 262144 + 32768);            // 1 MiB

    k_cvt_w<<<(D_ATT * D_ENC + 255) / 256, 256, 0, stream>>>(Wenc, wencb);
    k_dec_bias<<<BATCH, D_ATT, 0, stream>>>(dh, Wdec, bdec, benc, dbias);

    dim3 gs(T_IN / 64, BATCH);
    k_scores<<<gs, 128, 64 * LROW, stream>>>(enc, mask, wencb, dbias, vvec, wts);

    k_softmax<<<BATCH, 256, 0, stream>>>(wts);

    dim3 gc(16, BATCH);
    k_ctx_partial<<<gc, 256, 0, stream>>>(enc, wts, partial);
    k_ctx_reduce<<<(BATCH * D_ENC) / 256, 256, 0, stream>>>(partial, ctx);
}
